// Net_58729382805604
// MI455X (gfx1250) — matrix operand order repaired and checked by reading, not compiled, not run on hardware
//
#include <hip/hip_runtime.h>
#include <hip/hip_bf16.h>
#include <stdint.h>

#define N_NODES 100000
#define N_EDGES 1600000
#define D_FEAT  512
#define HIDDEN  64
#define N_CLASS 16

typedef __attribute__((ext_vector_type(16))) __bf16 v16bf;
typedef __attribute__((ext_vector_type(8)))  float  v8f;

// ---------------- init / degree / normalization ----------------

__global__ void init_ws(float* __restrict__ deg, float* __restrict__ agg1,
                        float* __restrict__ agg2) {
    int t = blockIdx.x * blockDim.x + threadIdx.x;
    if (t < N_NODES * HIDDEN)  agg1[t] = 0.0f;
    if (t < N_NODES)           deg[t]  = 1.0f;   // self-loop weight
    if (t < N_NODES * N_CLASS) agg2[t] = 0.0f;
}

__global__ void accum_deg(const int* __restrict__ dst, const float* __restrict__ w,
                          float* __restrict__ deg) {
    int e = blockIdx.x * blockDim.x + threadIdx.x;
    if (e < N_EDGES) atomicAdd(deg + dst[e], w[e]);
}

__global__ void deg_to_dis(float* __restrict__ deg) {
    int i = blockIdx.x * blockDim.x + threadIdx.x;
    if (i < N_NODES) {
        float d = deg[i];
        deg[i] = (d > 0.0f) ? rsqrtf(fmaxf(d, 1e-12f)) : 0.0f;
    }
}

// ---------------- weight pre-swizzle into B-operand register layout ----------------
// For each 32-wide K step and each output column, the 32 K values are contiguous:
//   Wsw[ ((k>>5)*COLS + col)*32 + (k&31) ] = bf16( W[k*COLS + col] )
// A lane (col=lane&15 within tile, half=lane>>4) then reads its 16-element v16bf
// B fragment (K = half*16 .. half*16+15) as ONE aligned 32-byte vector load.

__global__ void swizzleW1(const float* __restrict__ W, __bf16* __restrict__ Wsw) {
    int t = blockIdx.x * blockDim.x + threadIdx.x;
    if (t >= D_FEAT * HIDDEN) return;
    int k = t >> 6, col = t & 63;
    Wsw[((size_t)(k >> 5) * HIDDEN + col) * 32 + (k & 31)] = (__bf16)W[t];
}

__global__ void swizzleW2(const float* __restrict__ W, __bf16* __restrict__ Wsw) {
    int t = blockIdx.x * blockDim.x + threadIdx.x;
    if (t >= HIDDEN * N_CLASS) return;
    int k = t >> 4, col = t & 15;
    Wsw[((size_t)(k >> 5) * N_CLASS + col) * 32 + (k & 31)] = (__bf16)W[t];
}

// ---------------- GEMM1: [100000,512] x [512,64] -> h1, bf16 WMMA ----------------
// 8 waves/block; each wave owns 16 rows x ALL 64 cols (4 accumulators), so the
// A fragment is loaded once per K-step and feeds 4 WMMAs. Block = 128 rows.
__global__ __launch_bounds__(256) void gemm1(const float* __restrict__ X,
                                             const __bf16* __restrict__ Wsw,
                                             float* __restrict__ H) {
    const int lane = threadIdx.x & 31;
    const int wave = threadIdx.x >> 5;
    const int row0 = blockIdx.x * 128 + wave * 16;
    const int m    = lane & 15;
    const int half = lane >> 4;
    int ra = row0 + m; if (ra >= N_NODES) ra = N_NODES - 1;   // clamp: EXEC stays full
    const float* xrow = X + (size_t)ra * D_FEAT;

    v8f acc0 = {}, acc1 = {}, acc2 = {}, acc3 = {};
    for (int k0 = 0; k0 < D_FEAT; k0 += 32) {
        // A 16x32 bf16 fragment: a[j]=K(half*16+j), j=0..15
        v16bf a;
        const float* pa = xrow + k0 + half * 16;
#pragma unroll
        for (int j = 0; j < 8; ++j) a[j]     = (__bf16)pa[j];
#pragma unroll
        for (int j = 0; j < 8; ++j) a[8 + j] = (__bf16)pa[8 + j];

        // 4 pre-swizzled B fragments (col tiles 0..3), each one 32B vector load
        const __bf16* pb = Wsw + ((size_t)(k0 >> 5) * HIDDEN + m) * 32 + half * 16;
        v16bf b0 = *(const v16bf*)(pb + 0 * 16 * 32);
        v16bf b1 = *(const v16bf*)(pb + 1 * 16 * 32);
        v16bf b2 = *(const v16bf*)(pb + 2 * 16 * 32);
        v16bf b3 = *(const v16bf*)(pb + 3 * 16 * 32);

        acc0 = __builtin_amdgcn_wmma_f32_16x16x32_bf16(false, a, false, b0, (short)0, acc0, false, false);
        acc1 = __builtin_amdgcn_wmma_f32_16x16x32_bf16(false, a, false, b1, (short)0, acc1, false, false);
        acc2 = __builtin_amdgcn_wmma_f32_16x16x32_bf16(false, a, false, b2, (short)0, acc2, false, false);
        acc3 = __builtin_amdgcn_wmma_f32_16x16x32_bf16(false, a, false, b3, (short)0, acc3, false, false);
    }

    // D layout: VGPR r -> row half*8+r, col (ct*16 + m)
#pragma unroll
    for (int r = 0; r < 8; ++r) {
        int rr = row0 + half * 8 + r;
        if (rr < N_NODES) {
            float* out = H + (size_t)rr * HIDDEN + m;
            out[0]  = acc0[r];
            out[16] = acc1[r];
            out[32] = acc2[r];
            out[48] = acc3[r];
        }
    }
}

// ---------------- edge scatter, layer 1 (HIDDEN=64, 16 thr/edge x 4 cols) ----------------

__global__ void scatter1(const int* __restrict__ src, const int* __restrict__ dst,
                         const float* __restrict__ w, const float* __restrict__ dis,
                         const float* __restrict__ h, float* __restrict__ agg) {
    long long t = (long long)blockIdx.x * blockDim.x + threadIdx.x;
    long long e = t >> 4;
    if (e >= (long long)(N_EDGES + N_NODES)) return;
    int c = ((int)(t & 15)) * 4;
    int s, d; float ew;
    if (e < N_EDGES) { s = src[e]; d = dst[e]; ew = w[e]; }
    else             { s = d = (int)(e - N_EDGES); ew = 1.0f; }  // self-loop
    float nrm = dis[s] * ew * dis[d];
    const float4 hv = *(const float4*)(h + (size_t)s * HIDDEN + c);
    float* out = agg + (size_t)d * HIDDEN + c;
    atomicAdd(out + 0, hv.x * nrm);
    atomicAdd(out + 1, hv.y * nrm);
    atomicAdd(out + 2, hv.z * nrm);
    atomicAdd(out + 3, hv.w * nrm);
}

// ---------------- bias + ReLU + deterministic dropout (p=0.5) ----------------

__global__ void relu_drop(float* __restrict__ a, const float* __restrict__ b1) {
    int t = blockIdx.x * blockDim.x + threadIdx.x;
    if (t >= N_NODES * HIDDEN) return;
    float v = fmaxf(a[t] + b1[t & (HIDDEN - 1)], 0.0f);
    uint32_t h = (uint32_t)t * 2654435761u;
    h ^= h >> 16; h *= 0x45d9f3bu; h ^= h >> 13;
    a[t] = (h & 1u) ? v * 2.0f : 0.0f;             // keep-prob 0.5, scale 1/(1-p)
}

// ---------------- GEMM2: [100000,64] x [64,16] -> h2, bf16 WMMA ----------------
// 8 waves/block, one 16x16 tile each (N_CLASS is exactly one tile) -> 128 rows/block.
__global__ __launch_bounds__(256) void gemm2(const float* __restrict__ Hd,
                                             const __bf16* __restrict__ Wsw,
                                             float* __restrict__ O) {
    const int lane = threadIdx.x & 31;
    const int wave = threadIdx.x >> 5;
    const int row0 = blockIdx.x * 128 + wave * 16;
    const int m    = lane & 15;
    const int half = lane >> 4;
    int ra = row0 + m; if (ra >= N_NODES) ra = N_NODES - 1;
    const float* arow = Hd + (size_t)ra * HIDDEN;

    v8f acc = {};
#pragma unroll
    for (int k0 = 0; k0 < HIDDEN; k0 += 32) {
        v16bf a;
        const float* pa = arow + k0 + half * 16;
#pragma unroll
        for (int j = 0; j < 8; ++j) a[j]     = (__bf16)pa[j];
#pragma unroll
        for (int j = 0; j < 8; ++j) a[8 + j] = (__bf16)pa[8 + j];
        v16bf b = *(const v16bf*)(Wsw + ((size_t)(k0 >> 5) * N_CLASS + m) * 32 + half * 16);
        acc = __builtin_amdgcn_wmma_f32_16x16x32_bf16(
            false, a, false, b, (short)0, acc, false, false);
    }
#pragma unroll
    for (int r = 0; r < 8; ++r) {
        int rr = row0 + half * 8 + r;
        if (rr < N_NODES) O[(size_t)rr * N_CLASS + m] = acc[r];
    }
}

// ---------------- edge scatter, layer 2 (N_CLASS=16, 16 thr/edge) ----------------

__global__ void scatter2(const int* __restrict__ src, const int* __restrict__ dst,
                         const float* __restrict__ w, const float* __restrict__ dis,
                         const float* __restrict__ h, float* __restrict__ agg) {
    long long t = (long long)blockIdx.x * blockDim.x + threadIdx.x;
    long long e = t >> 4;
    if (e >= (long long)(N_EDGES + N_NODES)) return;
    int c = (int)(t & 15);
    int s, d; float ew;
    if (e < N_EDGES) { s = src[e]; d = dst[e]; ew = w[e]; }
    else             { s = d = (int)(e - N_EDGES); ew = 1.0f; }
    float nrm = dis[s] * ew * dis[d];
    atomicAdd(agg + (size_t)d * N_CLASS + c, h[(size_t)s * N_CLASS + c] * nrm);
}

// ---------------- bias + log_softmax; writes both outputs ----------------

__global__ void logsoftmax_out(const float* __restrict__ agg, const float* __restrict__ b2,
                               float* __restrict__ out) {
    int i = blockIdx.x * blockDim.x + threadIdx.x;
    if (i >= N_NODES) return;
    float v[N_CLASS];
    float mx = -3.402823466e+38f;
#pragma unroll
    for (int j = 0; j < N_CLASS; ++j) {
        v[j] = agg[(size_t)i * N_CLASS + j] + b2[j];
        mx = fmaxf(mx, v[j]);
    }
    float ssum = 0.0f;
#pragma unroll
    for (int j = 0; j < N_CLASS; ++j) ssum += __expf(v[j] - mx);
    float lse = mx + __logf(ssum);
#pragma unroll
    for (int j = 0; j < N_CLASS; ++j) {
        out[(size_t)i * N_CLASS + j] = v[j] - lse;                                  // log_softmax
        out[(size_t)N_NODES * N_CLASS + (size_t)i * N_CLASS + j] = v[j];            // x_out
    }
}

// ---------------- launch ----------------

extern "C" void kernel_launch(void* const* d_in, const int* in_sizes, int n_in,
                              void* d_out, int out_size, void* d_ws, size_t ws_size,
                              hipStream_t stream) {
    const float* x    = (const float*)d_in[0];
    const int*   eidx = (const int*)  d_in[1];   // [2, N_EDGES]
    const float* ew   = (const float*)d_in[2];
    const float* W1   = (const float*)d_in[3];
    const float* b1   = (const float*)d_in[4];
    const float* W2   = (const float*)d_in[5];
    const float* b2   = (const float*)d_in[6];
    const int* src = eidx;
    const int* dst = eidx + N_EDGES;

    float* deg   = (float*)d_ws;                          // N          (becomes rsqrt-deg)
    float* h1    = deg  + N_NODES;                        // N*64
    float* agg1  = h1   + (size_t)N_NODES * HIDDEN;       // N*64
    float* agg2  = agg1 + (size_t)N_NODES * HIDDEN;       // N*16
    __bf16* wsw1 = (__bf16*)(agg2 + (size_t)N_NODES * N_CLASS);   // 512*64 bf16 (64KB)
    __bf16* wsw2 = wsw1 + (size_t)D_FEAT * HIDDEN;                // 64*16 bf16 (2KB)
    float* h2    = h1;                                    // h1 dead once gemm2 starts
    float* out   = (float*)d_out;

    const int TB = 256;
    init_ws<<<(N_NODES * HIDDEN + TB - 1) / TB, TB, 0, stream>>>(deg, agg1, agg2);
    accum_deg<<<(N_EDGES + TB - 1) / TB, TB, 0, stream>>>(dst, ew, deg);
    deg_to_dis<<<(N_NODES + TB - 1) / TB, TB, 0, stream>>>(deg);

    swizzleW1<<<(D_FEAT * HIDDEN + TB - 1) / TB, TB, 0, stream>>>(W1, wsw1);
    swizzleW2<<<(HIDDEN * N_CLASS + TB - 1) / TB, TB, 0, stream>>>(W2, wsw2);

    gemm1<<<(N_NODES + 127) / 128, TB, 0, stream>>>(x, wsw1, h1);

    long long sc_threads = (long long)(N_EDGES + N_NODES) * 16;
    int sc_blocks = (int)((sc_threads + TB - 1) / TB);
    scatter1<<<sc_blocks, TB, 0, stream>>>(src, dst, ew, deg, h1, agg1);

    relu_drop<<<(N_NODES * HIDDEN + TB - 1) / TB, TB, 0, stream>>>(agg1, b1);

    gemm2<<<(N_NODES + 127) / 128, TB, 0, stream>>>(agg1, wsw2, h2);

    scatter2<<<sc_blocks, TB, 0, stream>>>(src, dst, ew, deg, h2, agg2);

    logsoftmax_out<<<(N_NODES + TB - 1) / TB, TB, 0, stream>>>(agg2, b2, out);
}
